// PtrNetwork_5763846111530
// MI455X (gfx1250) — matrix operand order repaired and checked by reading, not compiled, not run on hardware
//
#include <hip/hip_runtime.h>

typedef _Float16 v16h __attribute__((ext_vector_type(16)));
typedef _Float16 v8h  __attribute__((ext_vector_type(8)));
typedef float    v8f  __attribute__((ext_vector_type(8)));

#define D_MODEL 512
#define SRC_LEN 256
#define TGT_LEN 256
#define BATCH   8
#define TT      16   // t-tile in scores kernel

// ---- gfx1250 hardware tanh (TRANS32 op). Prefer the builtin so the compiler
// ---- handles TRANS hazards; otherwise inline asm + v_nop (ISA: 1 independent
// ---- op or V_NOP required after a TRANS op before its result is used).
#if defined(__has_builtin) && __has_builtin(__builtin_amdgcn_tanhf)
__device__ __forceinline__ float hw_tanhf(float x) { return __builtin_amdgcn_tanhf(x); }
#elif defined(__has_builtin) && __has_builtin(__builtin_amdgcn_tanh_f32)
__device__ __forceinline__ float hw_tanhf(float x) { return __builtin_amdgcn_tanh_f32(x); }
#else
__device__ __forceinline__ float hw_tanhf(float x) {
    float r;
    asm volatile("v_tanh_f32 %0, %1\n\tv_nop" : "=v"(r) : "v"(x));
    return r;
}
#endif

// ---------------- prep: f32 -> f16 cast ----------------
__global__ void k_f32_to_f16(const float* __restrict__ src, _Float16* __restrict__ dst, int n) {
    int i = blockIdx.x * blockDim.x + threadIdx.x;
    if (i < n) dst[i] = (_Float16)src[i];
}

// ---------------- prep: W[K=512][N=256] -> Wt[N][K] f16 ----------------
__global__ void k_transpose_to_f16(const float* __restrict__ W, _Float16* __restrict__ Wt) {
    int i = blockIdx.x * blockDim.x + threadIdx.x;   // i = n*512 + k
    int n = i >> 9, k = i & 511;
    Wt[i] = (_Float16)W[k * SRC_LEN + n];
}

// ---------------- WMMA GEMM: C[M=2048,N=256] = A[M,512] x Wt^T ----------------
// One wave per 16x16 C tile, K unrolled in steps of 32 (v_wmma_f32_16x16x32_f16).
// transposed==0: C[m*256+n] (dt layout). transposed==1: store et_t[b][n][s].
__global__ __launch_bounds__(128) void k_gemm_wmma(const _Float16* __restrict__ A,
                                                   const _Float16* __restrict__ Bt,
                                                   float* __restrict__ C, int transposed) {
    const int lane = threadIdx.x & 31;
    const int wave = threadIdx.x >> 5;
    const int tile = blockIdx.x * 4 + wave;      // 2048 tiles total
    const int nt = tile & 15;                    // N/16 = 16 tiles
    const int mt = tile >> 4;                    // M/16 = 128 tiles
    const int m0 = mt << 4, n0 = nt << 4;

    const int row = lane & 15;
    const int hi  = lane >> 4;                   // lane group 0/1
    // A fragment: lanes 0-15 hold K 0..15; lanes 16-31 hold K 16..31
    const _Float16* arow = A  + (size_t)(m0 + row) * D_MODEL + hi * 16;
    // B fragment: lanes 0-15 hold K 0..15; lanes 16-31 hold K 16..31 (contiguous in Wt[n][k])
    const _Float16* brow = Bt + (size_t)(n0 + row) * D_MODEL + hi * 16;

    v8f c = {};
#pragma unroll
    for (int k0 = 0; k0 < D_MODEL; k0 += 32) {
        v8h alo = *(const v8h*)(arow + k0);
        v8h ahi = *(const v8h*)(arow + k0 + 8);
        v16h a = __builtin_shufflevector(alo, ahi, 0,1,2,3,4,5,6,7,8,9,10,11,12,13,14,15);
        v16h b = *(const v16h*)(brow + k0);
        c = __builtin_amdgcn_wmma_f32_16x16x32_f16(false, a, false, b, (short)0, c, false, false);
    }

    const int n = n0 + row;
#pragma unroll
    for (int r = 0; r < 8; ++r) {
        int m = m0 + r + hi * 8;                 // C layout: VGPR r -> M = r (+8 for lanes>=16)
        if (!transposed) {
            C[(size_t)m * SRC_LEN + n] = c[r];   // dt[b*T + t][l]
        } else {
            int b_ = m >> 8, sl = m & 255;       // m = b*SRC + s
            C[((size_t)(b_ << 8) + n) * SRC_LEN + sl] = c[r];   // et_t[b][l][s]
        }
    }
}

// ---------------- scores + mask + log_softmax + transpose ----------------
__global__ __launch_bounds__(256) void k_scores(const float* __restrict__ dt,
                                                const float* __restrict__ et_t,
                                                const float* __restrict__ vt,
                                                const int* __restrict__ lens,
                                                float* __restrict__ out) {
    __shared__ float dts[SRC_LEN][TT];   // 16KB, transposed: [l][tt] -> per-l 64B contiguous
    __shared__ float vts[SRC_LEN];       // 1KB
    __shared__ float red[TT][8];         // reduction scratch (8 waves)

    const int b  = blockIdx.y;
    const int t0 = blockIdx.x * TT;
    const int s  = threadIdx.x;
    const int lane = s & 31, wave = s >> 5;

    // load dt tile (global reads coalesced in l; one-time strided LDS writes)
    for (int i = s; i < TT * SRC_LEN; i += 256) {
        int tt = i >> 8, l = i & 255;
        dts[l][tt] = dt[((size_t)(b * TGT_LEN) + t0 + tt) * SRC_LEN + l];
    }
    vts[s] = vt[s];
    __syncthreads();

    float acc[TT];
#pragma unroll
    for (int tt = 0; tt < TT; ++tt) acc[tt] = 0.f;

    const float* etb = et_t + (size_t)b * SRC_LEN * SRC_LEN;   // [l][s]
    for (int l = 0; l < SRC_LEN; ++l) {
        float e = etb[l * SRC_LEN + s];      // coalesced across threads
        float v = vts[l];                    // LDS broadcast
        float dl[TT];
        const float4* dp = (const float4*)&dts[l][0];   // 4x ds_load_b128 broadcasts
        ((float4*)dl)[0] = dp[0];
        ((float4*)dl)[1] = dp[1];
        ((float4*)dl)[2] = dp[2];
        ((float4*)dl)[3] = dp[3];
#pragma unroll
        for (int tt = 0; tt < TT; ++tt)
            acc[tt] += v * hw_tanhf(dl[tt] + e);
    }

    // mask: scores += log(mask + 1e-8)
    float madd = logf(((s < lens[b]) ? 1.0f : 0.0f) + 1e-8f);
#pragma unroll
    for (int tt = 0; tt < TT; ++tt) acc[tt] += madd;

    // log_softmax over s (256 threads) for each tt: max
#pragma unroll
    for (int tt = 0; tt < TT; ++tt) {
        float w = acc[tt];
        for (int o = 16; o > 0; o >>= 1) w = fmaxf(w, __shfl_xor(w, o, 32));
        if (lane == 0) red[tt][wave] = w;
    }
    __syncthreads();
#pragma unroll
    for (int tt = 0; tt < TT; ++tt) {
        float m = red[tt][0];
        for (int i = 1; i < 8; ++i) m = fmaxf(m, red[tt][i]);
        acc[tt] -= m;
    }
    __syncthreads();
    // sum of exp
#pragma unroll
    for (int tt = 0; tt < TT; ++tt) {
        float w = __expf(acc[tt]);
        for (int o = 16; o > 0; o >>= 1) w += __shfl_xor(w, o, 32);
        if (lane == 0) red[tt][wave] = w;
    }
    __syncthreads();
    float z[TT];
#pragma unroll
    for (int tt = 0; tt < TT; ++tt) {
        float sum = 0.f;
        for (int i = 0; i < 8; ++i) sum += red[tt][i];
        z[tt] = logf(sum);
    }

    // out[b][s][t0+tt] = acc - z   (transposed output [B, src, tgt])
    float* op = out + ((size_t)(b * SRC_LEN + s)) * TGT_LEN + t0;
#pragma unroll
    for (int q = 0; q < 4; ++q) {
        float4 o4;
        o4.x = acc[4*q+0] - z[4*q+0];
        o4.y = acc[4*q+1] - z[4*q+1];
        o4.z = acc[4*q+2] - z[4*q+2];
        o4.w = acc[4*q+3] - z[4*q+3];
        ((float4*)op)[q] = o4;
    }
}

extern "C" void kernel_launch(void* const* d_in, const int* in_sizes, int n_in,
                              void* d_out, int out_size, void* d_ws, size_t ws_size,
                              hipStream_t stream) {
    const float* dec  = (const float*)d_in[0];   // [8,256,512]
    const float* enc  = (const float*)d_in[1];   // [8,256,512]
    const int*   lens = (const int*)  d_in[2];   // [8]
    const float* W1   = (const float*)d_in[3];   // [512,256]
    const float* W2   = (const float*)d_in[4];   // [512,256]
    const float* vt   = (const float*)d_in[5];   // [256]
    float* out = (float*)d_out;                  // [8,256,256]

    char* ws = (char*)d_ws;
    float*    dt    = (float*)   (ws + 0);                       // 2 MB
    float*    et_t  = (float*)   (ws + (2u << 20));              // 2 MB  [b][l][s]
    _Float16* dec16 = (_Float16*)(ws + (4u << 20));              // 2 MB
    _Float16* enc16 = (_Float16*)(ws + (6u << 20));              // 2 MB
    _Float16* W1t   = (_Float16*)(ws + (8u << 20));              // 256 KB
    _Float16* W2t   = (_Float16*)(ws + (8u << 20) + (256u << 10)); // 256 KB

    const int nX = BATCH * 256 * D_MODEL;        // 1,048,576 elems each
    k_f32_to_f16<<<nX / 256, 256, 0, stream>>>(dec, dec16, nX);
    k_f32_to_f16<<<nX / 256, 256, 0, stream>>>(enc, enc16, nX);
    const int nW = D_MODEL * SRC_LEN;            // 131,072
    k_transpose_to_f16<<<nW / 256, 256, 0, stream>>>(W1, W1t);
    k_transpose_to_f16<<<nW / 256, 256, 0, stream>>>(W2, W2t);

    // 2048 16x16 tiles each, 4 waves (tiles) per 128-thread block
    k_gemm_wmma<<<512, 128, 0, stream>>>(dec16, W1t, dt, 0);
    k_gemm_wmma<<<512, 128, 0, stream>>>(enc16, W2t, et_t, 1);

    dim3 g(TGT_LEN / TT, BATCH);
    k_scores<<<g, 256, 0, stream>>>(dt, et_t, vt, lens, out);
}
